// KNNAttention_72129680769661
// MI455X (gfx1250) — compile-verified, matrix operand order checked by reading, not run on hardware
//
#include <hip/hip_runtime.h>
#include <hip/hip_bf16.h>
#include <cstdint>

#define BB 2
#define NN 2048
#define DIMM 512
#define HH 8
#define DHH 64
#define KK 32
#define NB (NN / 32)

typedef __attribute__((ext_vector_type(16))) __bf16 v16bf;
typedef __attribute__((ext_vector_type(8)))  __bf16 v8bf;
typedef __attribute__((ext_vector_type(8)))  float  v8f;
typedef __attribute__((ext_vector_type(4)))  int    v4i;

#ifndef __has_builtin
#define __has_builtin(x) 0
#endif
#if __has_builtin(__builtin_amdgcn_global_load_async_to_lds_b128)
#define HAS_ASYNC_LDS 1
#else
#define HAS_ASYNC_LDS 0
#endif
#if __has_builtin(__builtin_amdgcn_s_wait_asynccnt)
#define HAS_WAIT_ASYNC 1
#else
#define HAS_WAIT_ASYNC 0
#endif

#if HAS_ASYNC_LDS
static __device__ __forceinline__ void async_cp16(void* lds, const void* g) {
  __builtin_amdgcn_global_load_async_to_lds_b128(
      (__attribute__((address_space(1))) v4i*)g,
      (__attribute__((address_space(3))) v4i*)lds, 0, 0);
}
template <int N> static __device__ __forceinline__ void wait_async() {
#if HAS_WAIT_ASYNC
  __builtin_amdgcn_s_wait_asynccnt(N);
#else
  asm volatile("s_wait_asynccnt %0" ::"i"(N) : "memory");
#endif
}
#else
static __device__ __forceinline__ void async_cp16(void* lds, const void* g) {
  *reinterpret_cast<float4*>(lds) = *reinterpret_cast<const float4*>(g);
}
template <int N> static __device__ __forceinline__ void wait_async() {}
#endif

static __device__ __forceinline__ v8f wmma_bf16(v16bf a, v16bf b, v8f c) {
  return __builtin_amdgcn_wmma_f32_16x16x32_bf16(false, a, false, b, (short)0, c,
                                                 false, false);
}

static __device__ __forceinline__ v16bf cat8(v8bf lo, v8bf hi) {
  return __builtin_shufflevector(lo, hi, 0, 1, 2, 3, 4, 5, 6, 7, 8, 9, 10, 11,
                                 12, 13, 14, 15);
}
// Load the two contiguous 8-element K-chunks of a 16-bit WMMA operand:
// lane-half lh holds K = 8*lh..8*lh+7 and 16+8*lh..16+8*lh+7.
static __device__ __forceinline__ v16bf ld_op(const __bf16* row, int lh) {
  return cat8(*reinterpret_cast<const v8bf*>(row + 8 * lh),
              *reinterpret_cast<const v8bf*>(row + 16 + 8 * lh));
}

static __device__ __forceinline__ float rowmax16(float x) {
#pragma unroll
  for (int m = 1; m < 16; m <<= 1) x = fmaxf(x, __shfl_xor(x, m, 32));
  return x;
}
static __device__ __forceinline__ float rowsum16(float x) {
#pragma unroll
  for (int m = 1; m < 16; m <<= 1) x += __shfl_xor(x, m, 32);
  return x;
}

// ---------------------------------------------------------------------------
// Kernel 0: transpose + convert weights to bf16: Wt[col][k] = W[k][col]
// ---------------------------------------------------------------------------
__global__ __launch_bounds__(256) void prep_kernel(
    const float* __restrict__ Wq, const float* __restrict__ Wkv,
    const float* __restrict__ Wout, __bf16* __restrict__ Wqt,
    __bf16* __restrict__ Wkvt, __bf16* __restrict__ Woutt) {
  const int t = blockIdx.x * 256 + threadIdx.x;  // 0 .. 512*512-1
  const int k = t & (DIMM - 1), col = t >> 9;
  Wqt[(size_t)col * DIMM + k] = (__bf16)Wq[(size_t)k * DIMM + col];
  Woutt[(size_t)col * DIMM + k] = (__bf16)Wout[(size_t)k * DIMM + col];
  if (col < 2 * DHH)
    Wkvt[(size_t)col * DIMM + k] = (__bf16)Wkv[(size_t)k * (2 * DHH) + col];
}

// ---------------------------------------------------------------------------
// Kernel 1: q/k/v projections + l2norm. 10 waves: 0..7 -> q heads (f32 out),
// wave 8 -> k (bf16, row-major), wave 9 -> v (bf16, d-major).
// ---------------------------------------------------------------------------
__global__ __launch_bounds__(320) void proj_kernel(
    const float* __restrict__ x, const __bf16* __restrict__ Wqt,
    const __bf16* __restrict__ Wkvt, float* __restrict__ qn,
    __bf16* __restrict__ kbuf, __bf16* __restrict__ vtr) {
  __shared__ __attribute__((aligned(16))) __bf16 x_lds[16][DIMM];

  const int row0 = blockIdx.x * 16;  // flat row in [0, B*N)
  const int b = row0 >> 11;
  const int n0 = row0 & (NN - 1);
  const int tid = threadIdx.x;

  for (int t = tid; t < 16 * DIMM / 4; t += blockDim.x) {
    int r = t >> 7, c4 = (t & 127) * 4;
    float4 xv = *(const float4*)(x + (size_t)(row0 + r) * DIMM + c4);
    __bf16* dst = &x_lds[r][c4];
    dst[0] = (__bf16)xv.x; dst[1] = (__bf16)xv.y;
    dst[2] = (__bf16)xv.z; dst[3] = (__bf16)xv.w;
  }
  __syncthreads();

  const int wave = tid >> 5, lane = tid & 31;
  const int ln = lane & 15, lh = lane >> 4;

  const __bf16* wt;
  int cb;
  if (wave < 8) { wt = Wqt;  cb = wave * DHH; }
  else          { wt = Wkvt; cb = (wave - 8) * DHH; }

  float acc[4][8];
#pragma unroll
  for (int dt = 0; dt < 4; ++dt)
#pragma unroll
    for (int g = 0; g < 8; ++g) acc[dt][g] = 0.f;

  for (int kb = 0; kb < DIMM / 32; ++kb) {
    v16bf a = ld_op(&x_lds[ln][kb * 32], lh);
#pragma unroll
    for (int dt = 0; dt < 4; ++dt) {
      v16bf bm = ld_op(wt + (size_t)(cb + dt * 16 + ln) * DIMM + kb * 32, lh);
      v8f c;
#pragma unroll
      for (int g = 0; g < 8; ++g) c[g] = acc[dt][g];
      c = wmma_bf16(a, bm, c);
#pragma unroll
      for (int g = 0; g < 8; ++g) acc[dt][g] = c[g];
    }
  }

  if (wave < 9) {  // l2norm rows of q heads and of k
#pragma unroll
    for (int g = 0; g < 8; ++g) {
      float ss = 0.f;
#pragma unroll
      for (int dt = 0; dt < 4; ++dt) ss += acc[dt][g] * acc[dt][g];
      ss = rowsum16(ss);
      float inv = 1.0f / fmaxf(sqrtf(ss), 1e-12f);
#pragma unroll
      for (int dt = 0; dt < 4; ++dt) acc[dt][g] *= inv;
    }
  }

  if (wave < 8) {
#pragma unroll
    for (int g = 0; g < 8; ++g) {
      int n = n0 + g + 8 * lh;
#pragma unroll
      for (int dt = 0; dt < 4; ++dt)
        qn[(((size_t)b * HH + wave) * NN + n) * DHH + dt * 16 + ln] = acc[dt][g];
    }
  } else if (wave == 8) {
#pragma unroll
    for (int g = 0; g < 8; ++g) {
      int n = n0 + g + 8 * lh;
#pragma unroll
      for (int dt = 0; dt < 4; ++dt)
        kbuf[((size_t)b * NN + n) * DHH + dt * 16 + ln] = (__bf16)acc[dt][g];
    }
  } else {
#pragma unroll
    for (int g = 0; g < 8; ++g) {
      int n = n0 + g + 8 * lh;
#pragma unroll
      for (int dt = 0; dt < 4; ++dt)
        vtr[((size_t)b * DHH + dt * 16 + ln) * NN + n] = (__bf16)acc[dt][g];
    }
  }
}

// ---------------------------------------------------------------------------
// Kernel 2: attention. Fixed softmax max (l2norm bound) -> no online rescale.
// 4 waves x 16 query rows; async double-buffered k/v staging through LDS.
// ---------------------------------------------------------------------------
__global__ __launch_bounds__(128) void attn_kernel(
    const float* __restrict__ qn, const __bf16* __restrict__ kbuf,
    const __bf16* __restrict__ vtr, const float* __restrict__ mem_kv,
    const unsigned char* __restrict__ mem_mask,
    const float* __restrict__ scale_param, __bf16* __restrict__ attn_out) {
  __shared__ __attribute__((aligned(16))) float  q_lds[4][16][DHH];
  __shared__ __attribute__((aligned(16))) float  p_lds[4][16][32];
  __shared__ __attribute__((aligned(16))) __bf16 k_lds[2][32][DHH];  // [key][d]
  __shared__ __attribute__((aligned(16))) __bf16 v_lds[2][DHH][32];  // [d][key]

  const int b = blockIdx.z, h = blockIdx.y;
  const int wave = threadIdx.x >> 5, lane = threadIdx.x & 31;
  const int ln = lane & 15, lh = lane >> 4;
  const int row0 = (blockIdx.x * 4 + wave) * 16;

  const float scale = __expf(scale_param[h]);

  // stage this wave's q tile, with scale folded in
  const float* qsrc = qn + (((size_t)b * HH + h) * NN + row0) * DHH;
  for (int t = lane; t < 256; t += 32) {
    int r = t >> 4, c4 = (t & 15) * 4;
    float4 q4 = *(const float4*)(qsrc + (size_t)r * DHH + c4);
    q4.x *= scale; q4.y *= scale; q4.z *= scale; q4.w *= scale;
    *(float4*)&q_lds[wave][r][c4] = q4;
  }

  // loop-invariant WMMA A operand (scaled q, bf16), d-blocks 0 and 1
  v16bf qa0, qa1;
  {
    const float* qrow = &q_lds[wave][ln][0];
#pragma unroll
    for (int e = 0; e < 8; ++e) {
      qa0[e]     = (__bf16)qrow[8 * lh + e];
      qa0[8 + e] = (__bf16)qrow[16 + 8 * lh + e];
      qa1[e]     = (__bf16)qrow[32 + 8 * lh + e];
      qa1[8 + e] = (__bf16)qrow[48 + 8 * lh + e];
    }
  }

  // ---- memory-key phase: scores, fixed row max, p, and mem_v into O -------
  float sm[2][8];
#pragma unroll
  for (int t = 0; t < 2; ++t) {
#pragma unroll
    for (int g = 0; g < 8; ++g) {
      const int r = g + 8 * lh, gr = row0 + r;
      const int j = t * 16 + ln;
      const size_t kb0 = ((((size_t)b * HH + h) * NN + gr) * KK + j) * 2 * DHH;
      const float4* qv = (const float4*)&q_lds[wave][r][0];
      const float4* kv4 = (const float4*)(mem_kv + kb0);
      float d = 0.f;
#pragma unroll
      for (int q4 = 0; q4 < DHH / 4; ++q4) {
        float4 qa = qv[q4], ka = kv4[q4];
        d += qa.x * ka.x + qa.y * ka.y + qa.z * ka.z + qa.w * ka.w;
      }
      const bool msk = mem_mask[(((size_t)b * HH + h) * NN + gr) * KK + j] != 0;
      sm[t][g] = msk ? d : -3.0e38f;
    }
  }

  float row_m[8], lsum[8];
#pragma unroll
  for (int g = 0; g < 8; ++g) {
    // local scores are bounded by scale (|q_hat . k_hat| <= 1), so this max
    // is valid for the whole row -> no online rescaling needed.
    row_m[g] = fmaxf(rowmax16(fmaxf(sm[0][g], sm[1][g])), scale);
    float p0 = __expf(sm[0][g] - row_m[g]);
    float p1 = __expf(sm[1][g] - row_m[g]);
    lsum[g] = p0 + p1;
    p_lds[wave][g + 8 * lh][ln] = p0;
    p_lds[wave][g + 8 * lh][16 + ln] = p1;
  }

  // mem_v accumulation in lane-contiguous d-chunks (b128 loads), then
  // scatter to the WMMA C layout via the q LDS tile (q no longer needed).
#pragma unroll
  for (int g = 0; g < 8; ++g) {
    const int r = g + 8 * lh, gr = row0 + r;
    const float* prow = &p_lds[wave][r][0];
    const size_t v0 =
        ((((size_t)b * HH + h) * NN + gr) * KK) * 2 * DHH + DHH + 4 * ln;
    float4 o4 = {0.f, 0.f, 0.f, 0.f};
#pragma unroll
    for (int j = 0; j < KK; ++j) {
      float p = prow[j];
      float4 v4 = *(const float4*)(mem_kv + v0 + (size_t)j * 2 * DHH);
      o4.x += p * v4.x; o4.y += p * v4.y;
      o4.z += p * v4.z; o4.w += p * v4.w;
    }
    *(float4*)&q_lds[wave][r][4 * ln] = o4;
  }

  float o[4][8];
#pragma unroll
  for (int dt = 0; dt < 4; ++dt)
#pragma unroll
    for (int g = 0; g < 8; ++g)
      o[dt][g] = q_lds[wave][g + 8 * lh][dt * 16 + ln];

  // ---- local key loop: 32 keys/iter, double-buffered async staging --------
  const __bf16* kbase = kbuf + (size_t)b * NN * DHH;
  const __bf16* vbase = vtr + (size_t)b * DHH * NN;
  auto stage = [&](int buf, int kb) {
    for (int c = threadIdx.x; c < 256; c += 128) {
      // k block: one contiguous 4 KiB span
      async_cp16(&k_lds[buf][0][0] + c * 8,
                 kbase + (size_t)(kb * 32) * DHH + c * 8);
      // v block: 64 rows (d) x 64 B from d-major global
      int d = c >> 2, off = (c & 3) * 8;
      async_cp16(&v_lds[buf][d][off], vbase + (size_t)d * NN + kb * 32 + off);
    }
  };

  stage(0, 0);
  for (int kb = 0; kb < NB; ++kb) {
    const int cur = kb & 1;
    if (kb + 1 < NB) {
      stage(cur ^ 1, kb + 1);
      wait_async<4>();
    } else {
      wait_async<0>();
    }
    __syncthreads();

    // scores: S = q.k^T (16x32) = two 16x16 tiles, K-dim 64 = 2 wmmas each
    v8f s[2];
#pragma unroll
    for (int kh = 0; kh < 2; ++kh) {
      const __bf16* kr = &k_lds[cur][kh * 16 + ln][0];
      v16bf b0 = ld_op(kr, lh);
      v16bf b1 = ld_op(kr + 32, lh);
      v8f c = {};
      c = wmma_bf16(qa0, b0, c);
      c = wmma_bf16(qa1, b1, c);
      s[kh] = c;
    }

    // softmax numerators with fixed max; per-lane partial row sums
#pragma unroll
    for (int g = 0; g < 8; ++g) {
      float p0 = __expf(s[0][g] - row_m[g]);
      float p1 = __expf(s[1][g] - row_m[g]);
      lsum[g] += p0 + p1;
      p_lds[wave][g + 8 * lh][ln] = p0;
      p_lds[wave][g + 8 * lh][16 + ln] = p1;
    }

    // O += P(16x32) . V(32x64)
    v16bf pa;
    {
      const float* prow = &p_lds[wave][ln][0];
#pragma unroll
      for (int e = 0; e < 8; ++e) {
        pa[e]     = (__bf16)prow[8 * lh + e];
        pa[8 + e] = (__bf16)prow[16 + 8 * lh + e];
      }
    }
#pragma unroll
    for (int dt = 0; dt < 4; ++dt) {
      v16bf bv = ld_op(&v_lds[cur][dt * 16 + ln][0], lh);
      v8f c;
#pragma unroll
      for (int g = 0; g < 8; ++g) c[g] = o[dt][g];
      c = wmma_bf16(pa, bv, c);
#pragma unroll
      for (int g = 0; g < 8; ++g) o[dt][g] = c[g];
    }
    __syncthreads();
  }

  // epilogue: reduce row sums, normalize, store (b, n, h*64+d) bf16
#pragma unroll
  for (int g = 0; g < 8; ++g) {
    const int gr = row0 + g + 8 * lh;
    const float inv = 1.0f / rowsum16(lsum[g]);
#pragma unroll
    for (int dt = 0; dt < 4; ++dt)
      attn_out[(((size_t)b * NN + gr) * HH + h) * DHH + dt * 16 + ln] =
          (__bf16)(o[dt][g] * inv);
  }
}

// ---------------------------------------------------------------------------
// Kernel 3: out = attn_out(4096x512 bf16) @ Wout -> f32
// ---------------------------------------------------------------------------
__global__ __launch_bounds__(256) void outproj_kernel(
    const __bf16* __restrict__ ao, const __bf16* __restrict__ Woutt,
    float* __restrict__ out) {
  __shared__ __attribute__((aligned(16))) __bf16 a_lds[16][DIMM];
  const int row0 = blockIdx.x * 16;
  const int tid = threadIdx.x;

  // 16 KiB contiguous tile -> async copy straight into LDS
  const __bf16* src = ao + (size_t)row0 * DIMM;
  for (int t = tid; t < 1024; t += 256)
    async_cp16((char*)&a_lds[0][0] + t * 16, (const char*)src + t * 16);
  wait_async<0>();
  __syncthreads();

  const int wave = tid >> 5, lane = tid & 31;
  const int ln = lane & 15, lh = lane >> 4;
  const int cb = wave * 64;

  float acc[4][8];
#pragma unroll
  for (int dt = 0; dt < 4; ++dt)
#pragma unroll
    for (int g = 0; g < 8; ++g) acc[dt][g] = 0.f;

  for (int kb = 0; kb < DIMM / 32; ++kb) {
    v16bf a = ld_op(&a_lds[ln][kb * 32], lh);
#pragma unroll
    for (int dt = 0; dt < 4; ++dt) {
      v16bf bm = ld_op(Woutt + (size_t)(cb + dt * 16 + ln) * DIMM + kb * 32, lh);
      v8f c;
#pragma unroll
      for (int g = 0; g < 8; ++g) c[g] = acc[dt][g];
      c = wmma_bf16(a, bm, c);
#pragma unroll
      for (int g = 0; g < 8; ++g) acc[dt][g] = c[g];
    }
  }

#pragma unroll
  for (int g = 0; g < 8; ++g) {
    const int r = row0 + g + 8 * lh;
#pragma unroll
    for (int dt = 0; dt < 4; ++dt)
      out[(size_t)r * DIMM + cb + dt * 16 + ln] = acc[dt][g];
  }
}

// ---------------------------------------------------------------------------
extern "C" void kernel_launch(void* const* d_in, const int* in_sizes, int n_in,
                              void* d_out, int out_size, void* d_ws,
                              size_t ws_size, hipStream_t stream) {
  const float* x = (const float*)d_in[0];
  const float* mem_kv = (const float*)d_in[1];
  const unsigned char* mem_mask = (const unsigned char*)d_in[2];
  const float* Wq = (const float*)d_in[3];
  const float* Wkv = (const float*)d_in[4];
  const float* Wout = (const float*)d_in[5];
  const float* scale_param = (const float*)d_in[6];
  float* out = (float*)d_out;

  char* ws = (char*)d_ws;
  float*  qn    = (float*)ws;                                   // 8 MiB
  __bf16* kbuf  = (__bf16*)(ws + ((size_t)8 << 20));            // 512 KiB
  __bf16* vtr   = (__bf16*)(ws + ((size_t)8 << 20) + (512u << 10));
  __bf16* ao    = (__bf16*)(ws + ((size_t)9 << 20));            // 4 MiB
  __bf16* Wqt   = (__bf16*)(ws + ((size_t)13 << 20));           // 512 KiB
  __bf16* Woutt = (__bf16*)(ws + ((size_t)13 << 20) + (512u << 10));
  __bf16* Wkvt  = (__bf16*)(ws + ((size_t)14 << 20));           // 128 KiB

  prep_kernel<<<dim3((DIMM * DIMM) / 256), dim3(256), 0, stream>>>(
      Wq, Wkv, Wout, Wqt, Wkvt, Woutt);
  proj_kernel<<<dim3((BB * NN) / 16), dim3(320), 0, stream>>>(x, Wqt, Wkvt, qn,
                                                              kbuf, vtr);
  attn_kernel<<<dim3(NN / 64, HH, BB), dim3(128), 0, stream>>>(
      qn, kbuf, vtr, mem_kv, mem_mask, scale_param, ao);
  outproj_kernel<<<dim3((BB * NN) / 16), dim3(256), 0, stream>>>(ao, Woutt,
                                                                 out);
  (void)in_sizes; (void)n_in; (void)out_size; (void)ws_size;
}
